// PointRendMaskHead_17016660427471
// MI455X (gfx1250) — compile-verified, matrix operand order checked by reading, not run on hardware
//
#include <hip/hip_runtime.h>
#include <hip/hip_bf16.h>

// ---------------------------------------------------------------------------
// PointRend mask head for MI455X (gfx1250, wave32, WMMA + TDM).
// GEMMs: bf16 x bf16 -> f32 via v_wmma_f32_16x16x32_bf16.
// Tile staging: Tensor Data Mover (tensor_load_to_lds) double-buffered,
// with a vectorized manual path if the builtin is unavailable.
// ---------------------------------------------------------------------------

typedef __attribute__((ext_vector_type(16))) __bf16 v16bf;
typedef __attribute__((ext_vector_type(8)))  float  v8f;
typedef uint32_t u32x4  __attribute__((ext_vector_type(4)));
typedef int      i32x4v __attribute__((ext_vector_type(4)));
typedef int      i32x8v __attribute__((ext_vector_type(8)));

#if __has_builtin(__builtin_amdgcn_tensor_load_to_lds)
#define USE_TDM 1
#else
#define USE_TDM 0
#endif

#define R_BOX   64
#define NCLS    80
#define FH      256
#define FW      256
#define SIDE    14
#define OUTS    7
#define NPTS    784
#define KPH     352      // point-head K padded (336 -> 352, multiple of 32)

static constexpr int BM  = 64;
static constexpr int BN  = 64;
static constexpr int BK  = 32;
static constexpr int LDT = 40;   // LDS tile row stride in elems (64B data + 16B pad)

// ----------------------------- bilinear taps -------------------------------
__device__ __forceinline__ void make_taps(float xn, float yn, int W, int H,
                                          int off[4], float w[4]) {
  float x  = xn * (float)W - 0.5f;
  float y  = yn * (float)H - 0.5f;
  float x0 = floorf(x), y0 = floorf(y);
  float wx = x - x0,    wy = y - y0;
#pragma unroll
  for (int t = 0; t < 4; ++t) {
    int   dx = t & 1, dy = t >> 1;
    float xi = x0 + (float)dx;
    float yi = y0 + (float)dy;
    bool  valid = (xi >= 0.f) && (xi < (float)W) && (yi >= 0.f) && (yi < (float)H);
    int xc = (int)fminf(fmaxf(xi, 0.f), (float)(W - 1));
    int yc = (int)fminf(fmaxf(yi, 0.f), (float)(H - 1));
    float ww = (dx ? wx : 1.f - wx) * (dy ? wy : 1.f - wy);
    off[t] = yc * W + xc;
    w[t]   = valid ? ww : 0.f;
  }
}

// --------------------- TDM 2D tile load (global -> LDS) --------------------
// D# per CDNA5 ISA ch.8: group0 = {flags, lds_addr, global_addr, type=2},
// group1 = {data_size=2B + LDS pad (16B per 64B), tensor dims, tile dims,
// dim0 stride}. OOB rows/cols read as zero (handles N edges natively).
__device__ __forceinline__ void tdm_load_2d(uint32_t lds_off, const __bf16* gptr,
                                            uint32_t tile_w, uint32_t tile_h,
                                            uint32_t tens_w, uint32_t tens_h,
                                            uint64_t stride_elems) {
#if USE_TDM
  uint64_t ga = (uint64_t)(uintptr_t)gptr;
  u32x4 g0;
  g0[0] = 1u;                                            // count=1, user desc
  g0[1] = lds_off;                                       // lds_addr
  g0[2] = (uint32_t)ga;                                  // global_addr[31:0]
  g0[3] = (uint32_t)((ga >> 32) & 0x01ffffffu) | (2u << 30);  // [56:32] | type=2
  i32x8v g1;
  g1[0] = (int)((1u << 16) | (1u << 20) | (3u << 22) | (3u << 25));
  //            data_size=2B  pad_en      16 dw period  pad 4 dwords
  g1[1] = (int)((tens_w & 0xffffu) << 16);               // tensor_dim0[15:0]
  g1[2] = (int)((tens_w >> 16) | ((tens_h & 0xffffu) << 16));
  g1[3] = (int)((tens_h >> 16) | ((tile_w & 0xffffu) << 16));
  g1[4] = (int)(tile_h & 0xffffu);                       // tile_dim1 (tile_dim2=0)
  g1[5] = (int)(uint32_t)stride_elems;                   // dim0 stride[31:0]
  g1[6] = (int)(uint32_t)((stride_elems >> 32) & 0xffffu);
  g1[7] = 0;
  i32x4v z4 = {0, 0, 0, 0};
#if __clang_major__ >= 23
  i32x8v z8 = {0, 0, 0, 0, 0, 0, 0, 0};
  __builtin_amdgcn_tensor_load_to_lds(g0, g1, z4, z4, z8, 0);
#else
  __builtin_amdgcn_tensor_load_to_lds(g0, g1, z4, z4, 0);
#endif
#else
  (void)lds_off; (void)gptr; (void)tile_w; (void)tile_h;
  (void)tens_w; (void)tens_h; (void)stride_elems;
#endif
}

// ------------------------- weight repack (cast only) -----------------------
// Weights stay in their native [N][Ksrc] row-major layout = B^T operand.
// Wb[n][k] (ld = Kpad), cols Ksrc..Kpad zero-filled.
__global__ void k_wcast(const float* __restrict__ W, __bf16* __restrict__ Wb,
                        int N, int Ksrc, int Kpad) {
  long i = (long)blockIdx.x * blockDim.x + threadIdx.x;
  if (i >= (long)N * Kpad) return;
  long n = i / Kpad, k = i % Kpad;
  Wb[i] = (k < Ksrc) ? (__bf16)W[n * (long)Ksrc + k] : (__bf16)0.0f;
}

// ------------------------ bf16 WMMA GEMM (wave32) --------------------------
// C[M,N] = A[M,K] * Bt[N,K]^T (+bias, relu). M % 64 == 0, K % 32 == 0.
// out_mode 1: conv scatter  out[(row/49)*12544 + col*49 + row%49].
template <bool OUT_BF16>
__global__ __launch_bounds__(256)
void gemm_wmma(const __bf16* __restrict__ A, const __bf16* __restrict__ Bt,
               const float* __restrict__ bias, void* __restrict__ Cout,
               int M, int N, int K, int ldA, int ldB, int ldC,
               int relu, int out_mode) {
  __shared__ __align__(16) __bf16 As[2][BM][LDT];
  __shared__ __align__(16) __bf16 Bs[2][BN][LDT];

  const int tid   = threadIdx.x;
  const int lane  = tid & 31;
  const int wave  = tid >> 5;          // 0..7
  const int wm    = wave & 3;          // 16-row tile within 64
  const int wn    = wave >> 2;         // 32-col half within 64
  const long bm   = (long)blockIdx.x * BM;
  const long bn   = (long)blockIdx.y * BN;
  const int mrow  = lane & 15;         // row-in-tile (A: m, B/C: n)
  const int khalf = lane >> 4;         // which 16B K-chunk this lane owns

  v8f acc0 = {}; v8f acc1 = {};

  auto stage = [&](int bufi, int kk) {
#if USE_TDM
    if (wave == 0) {
      tdm_load_2d((uint32_t)(uintptr_t)&As[bufi][0][0], A + bm * (long)ldA + kk,
                  BK, BM, (uint32_t)(K - kk), (uint32_t)(M - bm), (uint64_t)ldA);
      tdm_load_2d((uint32_t)(uintptr_t)&Bs[bufi][0][0], Bt + bn * (long)ldB + kk,
                  BK, BN, (uint32_t)(K - kk), (uint32_t)(N - bn), (uint64_t)ldB);
    }
#else
    const int row = tid >> 2;
    const int kc  = (tid & 3) * 8;
    *(uint4*)&As[bufi][row][kc] =
        *(const uint4*)(A + (bm + row) * (long)ldA + kk + kc);
    uint4 bv = {0, 0, 0, 0};
    if (bn + row < N)
      bv = *(const uint4*)(Bt + (bn + row) * (long)ldB + kk + kc);
    *(uint4*)&Bs[bufi][row][kc] = bv;
#endif
  };
  auto wait_stage = [&]() {
#if USE_TDM
    if (wave == 0) __builtin_amdgcn_s_wait_tensorcnt(0);
#endif
    __syncthreads();
  };

  union FragU { v16bf v; uint4 q[2]; };

  stage(0, 0);
  wait_stage();

  const int nsteps = K / BK;
  for (int s = 0; s < nsteps; ++s) {
    const int buf = s & 1;
    if (s + 1 < nsteps) stage(buf ^ 1, (s + 1) * BK);   // overlap DMA w/ math

    // ---- fragments: two contiguous 16B chunks per lane -> ds_load_b128 ----
    FragU ua, ub0, ub1;
    const __bf16* arow  = &As[buf][16 * wm + mrow][0];
    const __bf16* brow0 = &Bs[buf][32 * wn + mrow][0];
    const __bf16* brow1 = &Bs[buf][32 * wn + 16 + mrow][0];
    ua.q[0]  = *(const uint4*)(arow + khalf * 8);
    ua.q[1]  = *(const uint4*)(arow + 16 + khalf * 8);
    ub0.q[0] = *(const uint4*)(brow0 + khalf * 8);
    ub0.q[1] = *(const uint4*)(brow0 + 16 + khalf * 8);
    ub1.q[0] = *(const uint4*)(brow1 + khalf * 8);
    ub1.q[1] = *(const uint4*)(brow1 + 16 + khalf * 8);

    acc0 = __builtin_amdgcn_wmma_f32_16x16x32_bf16(false, ua.v, false, ub0.v,
                                                   (short)0, acc0, false, false);
    acc1 = __builtin_amdgcn_wmma_f32_16x16x32_bf16(false, ua.v, false, ub1.v,
                                                   (short)0, acc1, false, false);
    wait_stage();
  }

  // ---- epilogue: C layout VGPR j -> M = j + 8*khalf, col = lane&15 ----
  const long col0 = bn + 32 * wn + mrow;
  const long col1 = col0 + 16;
#pragma unroll
  for (int j = 0; j < 8; ++j) {
    const long row = bm + 16 * wm + j + 8 * khalf;
    float v0 = acc0[j];
    float v1 = acc1[j];
#pragma unroll
    for (int half = 0; half < 2; ++half) {
      long  col = half ? col1 : col0;
      float v   = half ? v1 : v0;
      if (col < N) {
        v += bias[col];
        if (relu) v = fmaxf(v, 0.f);
        long off;
        if (out_mode == 1) {
          long r = row / 49, p = row % 49;
          off = r * 12544 + col * 49 + p;
        } else {
          off = row * (long)ldC + col;
        }
        if (OUT_BF16) ((__bf16*)Cout)[off] = (__bf16)v;
        else          ((float*)Cout)[off]  = v;
      }
    }
  }
}

// -------------------- 14x14 grid sample fused with im2col ------------------
__global__ void k_sample_grid(const float* __restrict__ fmap,
                              const float* __restrict__ boxes,
                              __bf16* __restrict__ Aconv) {
  int b = blockIdx.x;                 // r*196 + s
  int r = b / 196, s = b % 196;
  int sy = s / SIDE, sx = s % SIDE;
  float bx0 = boxes[r * 4 + 0], by0 = boxes[r * 4 + 1];
  float bx1 = boxes[r * 4 + 2], by1 = boxes[r * 4 + 3];
  float gx = ((float)sx + 0.5f) / (float)SIDE;
  float gy = ((float)sy + 0.5f) / (float)SIDE;
  float xn = (bx0 + gx * (bx1 - bx0)) * 0.25f / (float)FW;
  float yn = (by0 + gy * (by1 - by0)) * 0.25f / (float)FH;
  int off[4]; float w[4];
  make_taps(xn, yn, FW, FH, off, w);
  int c = threadIdx.x;                // 256 channels
  const float* img = fmap + (long)c * (FH * FW);
  float acc = img[off[0]] * w[0] + img[off[1]] * w[1] +
              img[off[2]] * w[2] + img[off[3]] * w[3];
  int oy = sy >> 1, ky = sy & 1, ox = sx >> 1, kx = sx & 1;
  long row = (long)r * 49 + oy * 7 + ox;
  long col = (long)c * 4 + ky * 2 + kx;
  Aconv[row * 1024 + col] = (__bf16)acc;
}

// ------------------------- bilinear 2x upsample ----------------------------
__global__ void k_upsample2x(const float* __restrict__ in, float* __restrict__ out,
                             int h, int w) {
  long rc = blockIdx.x;               // R*NCLS planes
  const float* src = in + rc * (long)(h * w);
  float* dst = out + rc * (long)(4 * h * w);
  int W = 2 * w, H = 2 * h;
  for (int i = threadIdx.x; i < H * W; i += blockDim.x) {
    int oy = i / W, ox = i % W;
    float sx = ((float)ox + 0.5f) * 0.5f - 0.5f;
    float sy = ((float)oy + 0.5f) * 0.5f - 0.5f;
    float x0f = floorf(sx), y0f = floorf(sy);
    float wx = sx - x0f, wy = sy - y0f;
    int x0 = (int)x0f, y0 = (int)y0f;
    int x0c = min(max(x0, 0), w - 1), x1c = min(max(x0 + 1, 0), w - 1);
    int y0c = min(max(y0, 0), h - 1), y1c = min(max(y0 + 1, 0), h - 1);
    dst[i] = (1.f - wx) * (1.f - wy) * src[y0c * w + x0c] +
             wx * (1.f - wy) * src[y0c * w + x1c] +
             (1.f - wx) * wy * src[y1c * w + x0c] +
             wx * wy * src[y1c * w + x1c];
  }
}

// ---------------- stratified uncertainty point selection -------------------
__global__ void k_select(const float* __restrict__ mask,
                         const int* __restrict__ pred_classes,
                         int* __restrict__ idx, int H, int W) {
  int r = blockIdx.x;
  int cls = pred_classes[r];
  int HW = H * W;
  int gs = HW / NPTS;
  const float* row = mask + ((long)r * NCLS + cls) * (long)HW;
  for (int p = threadIdx.x; p < NPTS; p += blockDim.x) {
    int base = p * gs;
    float best = -1e30f; int bi = base;
    for (int j = 0; j < gs; ++j) {
      float u = -fabsf(row[base + j]);
      if (u > best) { best = u; bi = base + j; }
    }
    idx[r * NPTS + p] = bi;
  }
}

// ---------- fine + coarse point sampling into padded hcat (K=352) ----------
__global__ void k_fill_hcat(const float* __restrict__ fmap,
                            const float* __restrict__ boxes,
                            const float* __restrict__ coarse,
                            const int* __restrict__ idx,
                            __bf16* __restrict__ hA, __bf16* __restrict__ hB,
                            int W, int H) {
  int b = blockIdx.x;                 // r*NPTS + p
  int r = b / NPTS;
  int ix = idx[b];
  float px = ((float)(ix % W) + 0.5f) / (float)W;
  float py = ((float)(ix / W) + 0.5f) / (float)H;
  float bx0 = boxes[r * 4 + 0], by0 = boxes[r * 4 + 1];
  float bx1 = boxes[r * 4 + 2], by1 = boxes[r * 4 + 3];
  float xn = (bx0 + px * (bx1 - bx0)) * 0.25f / (float)FW;
  float yn = (by0 + py * (by1 - by0)) * 0.25f / (float)FH;
  int off[4]; float w[4];
  make_taps(xn, yn, FW, FH, off, w);
  int c = threadIdx.x;                // 256 threads
  const float* img = fmap + (long)c * (FH * FW);
  float acc = img[off[0]] * w[0] + img[off[1]] * w[1] +
              img[off[2]] * w[2] + img[off[3]] * w[3];
  long base = (long)b * KPH;
  hA[base + c] = (__bf16)acc;
  if (c < NCLS) {                     // coarse point sample (7x7 logits)
    int o2[4]; float w2[4];
    make_taps(px, py, OUTS, OUTS, o2, w2);
    const float* cl = coarse + ((long)r * NCLS + c) * 49;
    float cv = cl[o2[0]] * w2[0] + cl[o2[1]] * w2[1] +
               cl[o2[2]] * w2[2] + cl[o2[3]] * w2[3];
    __bf16 bv = (__bf16)cv;
    hA[base + 256 + c] = bv;
    hB[base + 256 + c] = bv;
  }
  if (c >= 96 && c < 112) {           // K padding cols 336..351
    hA[base + 240 + c] = (__bf16)0.0f;
    hB[base + 240 + c] = (__bf16)0.0f;
  }
}

// ------------------------- scatter refined logits --------------------------
__global__ void k_scatter(const float* __restrict__ plog,
                          const int* __restrict__ idx,
                          float* __restrict__ mask, int HW) {
  int b = blockIdx.x;                 // r*NPTS + p
  int r = b / NPTS;
  int ix = idx[b];
  int c = threadIdx.x;
  if (c < NCLS)
    mask[((long)r * NCLS + c) * (long)HW + ix] = plog[(long)b * NCLS + c];
}

// ===========================================================================
extern "C" void kernel_launch(void* const* d_in, const int* in_sizes, int n_in,
                              void* d_out, int out_size, void* d_ws, size_t ws_size,
                              hipStream_t stream) {
  const float* features = (const float*)d_in[0];
  const float* boxes    = (const float*)d_in[1];
  const int*   pclass   = (const int*)d_in[2];
  const float* conv_w   = (const float*)d_in[3];
  const float* conv_b   = (const float*)d_in[4];
  const float* fc1_w    = (const float*)d_in[5];
  const float* fc1_b    = (const float*)d_in[6];
  const float* fc2_w    = (const float*)d_in[7];
  const float* fc2_b    = (const float*)d_in[8];
  const float* cpred_w  = (const float*)d_in[9];
  const float* cpred_b  = (const float*)d_in[10];
  const float* pfc1_w   = (const float*)d_in[11];
  const float* pfc1_b   = (const float*)d_in[12];
  const float* pfc2_w   = (const float*)d_in[13];
  const float* pfc2_b   = (const float*)d_in[14];
  const float* pfc3_w   = (const float*)d_in[15];
  const float* pfc3_b   = (const float*)d_in[16];
  const float* ppred_w  = (const float*)d_in[17];
  const float* ppred_b  = (const float*)d_in[18];
  float* out = (float*)d_out;

  // bump allocator on workspace
  char* wsp = (char*)d_ws;
  auto alloc = [&](size_t bytes) -> void* {
    void* p = (void*)wsp;
    wsp += (bytes + 255) & ~(size_t)255;
    return p;
  };
  __bf16* Wconv = (__bf16*)alloc((size_t)256 * 1024 * 2);    // [256][1024]
  __bf16* Wfc1  = (__bf16*)alloc((size_t)1024 * 12544 * 2);  // [1024][12544]
  __bf16* Wfc2  = (__bf16*)alloc((size_t)1024 * 1024 * 2);   // [1024][1024]
  __bf16* Wcp   = (__bf16*)alloc((size_t)3920 * 1024 * 2);   // [3920][1024]
  __bf16* Wp1   = (__bf16*)alloc((size_t)256 * KPH * 2);     // [256][352]
  __bf16* Wp2   = (__bf16*)alloc((size_t)256 * KPH * 2);
  __bf16* Wp3   = (__bf16*)alloc((size_t)256 * KPH * 2);
  __bf16* Wpp   = (__bf16*)alloc((size_t)80 * KPH * 2);      // [80][352]
  __bf16* Aconv = (__bf16*)alloc((size_t)3136 * 1024 * 2);
  __bf16* Afc1  = (__bf16*)alloc((size_t)64 * 12544 * 2);
  __bf16* Aact1 = (__bf16*)alloc((size_t)64 * 1024 * 2);
  __bf16* Aact2 = (__bf16*)alloc((size_t)64 * 1024 * 2);
  float*  coarse= (float*)alloc((size_t)64 * 3920 * 4);
  float*  m14   = (float*)alloc((size_t)64 * NCLS * 196 * 4);
  float*  m28   = (float*)alloc((size_t)64 * NCLS * 784 * 4);
  float*  m56   = (float*)alloc((size_t)64 * NCLS * 3136 * 4);
  __bf16* hA    = (__bf16*)alloc((size_t)R_BOX * NPTS * KPH * 2);
  __bf16* hB    = (__bf16*)alloc((size_t)R_BOX * NPTS * KPH * 2);
  float*  plog  = (float*)alloc((size_t)R_BOX * NPTS * 80 * 4);
  int*    idx   = (int*)alloc((size_t)R_BOX * NPTS * 4);

  auto gemm = [&](const __bf16* A, const __bf16* Bt, const float* bias, void* C,
                  int M, int N, int K, int ldA, int ldB, int ldC,
                  int relu, int out_mode, bool outbf) {
    dim3 g(M / BM, (N + BN - 1) / BN, 1);
    if (outbf)
      gemm_wmma<true><<<g, 256, 0, stream>>>(A, Bt, bias, C, M, N, K, ldA, ldB, ldC, relu, out_mode);
    else
      gemm_wmma<false><<<g, 256, 0, stream>>>(A, Bt, bias, C, M, N, K, ldA, ldB, ldC, relu, out_mode);
  };

  // ---- weight repack to bf16 (dtype cast; native [N][K] layout kept) ----
  auto wcast = [&](const float* W, __bf16* Wb, int N, int Ksrc, int Kpad) {
    long total = (long)N * Kpad;
    k_wcast<<<(unsigned)((total + 255) / 256), 256, 0, stream>>>(W, Wb, N, Ksrc, Kpad);
  };
  wcast(conv_w,  Wconv, 256, 1024, 1024);    // [O][I*2*2] is im2col order already
  wcast(fc1_w,   Wfc1, 1024, 12544, 12544);
  wcast(fc2_w,   Wfc2, 1024, 1024, 1024);
  wcast(cpred_w, Wcp,  3920, 1024, 1024);
  wcast(pfc1_w,  Wp1,  256, 336, KPH);
  wcast(pfc2_w,  Wp2,  256, 336, KPH);
  wcast(pfc3_w,  Wp3,  256, 336, KPH);
  wcast(ppred_w, Wpp,  80, 336, KPH);

  // ---- coarse mask head ----
  k_sample_grid<<<R_BOX * 196, 256, 0, stream>>>(features, boxes, Aconv);
  gemm(Aconv, Wconv, conv_b, Afc1, 3136, 256, 1024, 1024, 1024, 0, 1, 1, true);
  gemm(Afc1, Wfc1, fc1_b, Aact1, 64, 1024, 12544, 12544, 12544, 1024, 1, 0, true);
  gemm(Aact1, Wfc2, fc2_b, Aact2, 64, 1024, 1024, 1024, 1024, 1024, 1, 0, true);
  gemm(Aact2, Wcp, cpred_b, coarse, 64, 3920, 1024, 1024, 1024, 3920, 0, 0, false);

  // ---- subdivision refinement ----
  auto refine = [&](float* mask, int H, int W) {
    int HW = H * W;
    k_select<<<R_BOX, 256, 0, stream>>>(mask, pclass, idx, H, W);
    k_fill_hcat<<<R_BOX * NPTS, 256, 0, stream>>>(features, boxes, coarse, idx, hA, hB, W, H);
    gemm(hA, Wp1, pfc1_b, hB, R_BOX * NPTS, 256, KPH, KPH, KPH, KPH, 1, 0, true);
    gemm(hB, Wp2, pfc2_b, hA, R_BOX * NPTS, 256, KPH, KPH, KPH, KPH, 1, 0, true);
    gemm(hA, Wp3, pfc3_b, hB, R_BOX * NPTS, 256, KPH, KPH, KPH, KPH, 1, 0, true);
    gemm(hB, Wpp, ppred_b, plog, R_BOX * NPTS, 80, KPH, KPH, KPH, 80, 0, 0, false);
    k_scatter<<<R_BOX * NPTS, 128, 0, stream>>>(plog, idx, mask, HW);
  };

  const int PLANES = R_BOX * NCLS;    // 5120
  k_upsample2x<<<PLANES, 256, 0, stream>>>(coarse, m14, 7, 7);    // step 0
  k_upsample2x<<<PLANES, 256, 0, stream>>>(m14, m28, 14, 14);     // step 1
  refine(m28, 28, 28);
  k_upsample2x<<<PLANES, 256, 0, stream>>>(m28, m56, 28, 28);     // step 2
  refine(m56, 56, 56);
  k_upsample2x<<<PLANES, 256, 0, stream>>>(m56, out, 56, 56);     // step 3
  refine(out, 112, 112);

  (void)in_sizes; (void)n_in; (void)out_size; (void)ws_size;
}
